// SrlBertOld_44856638440120
// MI455X (gfx1250) — matrix operand order repaired and checked by reading, not compiled, not run on hardware
//
#include <hip/hip_runtime.h>
#include <hip/hip_bf16.h>

// ---------------------------------------------------------------------------
// SRL-BERT biaffine head, fused for MI455X (gfx1250, wave32, WMMA f16->f32)
//   out[b,d_out,i,j] = tanh( sum_h x_i[h] x_j[h] W1[h,:] + xW2_i + xW3_j + bp ) @ W_out + bo
// GEMM1 (309 GFLOP) on v_wmma_f32_16x16x32_f16 with a 16(M) x 64(N) register
// tile per wave: the x_i*x_j A-fragment (4 loads + 8 v_pk_mul_f16) is reused
// by 4 WMMAs -> 3 b128 loads per WMMA instead of 6.
// ---------------------------------------------------------------------------

#define Hdim 768
#define NLAB 64
#define Bsz  4
#define Lseq 256

typedef __attribute__((ext_vector_type(16))) _Float16 v16h;
typedef __attribute__((ext_vector_type(8)))  _Float16 v8h;
typedef __attribute__((ext_vector_type(8)))  float    v8f;

union AF { v16h v; v8h h[2]; };

__device__ __forceinline__ v8h ld8(const _Float16* p) { return *(const v8h*)p; }

// tanh via exp2 (transcendental unit), saturates correctly for large |x|
__device__ __forceinline__ float ftanh(float x) {
  float ax = __builtin_fabsf(x);
  float e  = __builtin_amdgcn_exp2f(ax * 2.8853900817779268f);   // 2*log2(e)
  float r  = 1.0f - 2.0f * __builtin_amdgcn_rcpf(e + 1.0f);
  return __builtin_copysignf(r, x);
}

// ---------------- phase 0: pack / precompute ----------------

__global__ void pack_xh(const float* __restrict__ x, _Float16* __restrict__ xh, int n) {
  int i = blockIdx.x * blockDim.x + threadIdx.x;
  if (i < n) xh[i] = (_Float16)x[i];
}

// w1t[d][h] = W_proj[h][d]   (first H rows of W_proj)
__global__ void pack_w1t(const float* __restrict__ Wp, _Float16* __restrict__ w1t) {
  int i = blockIdx.x * blockDim.x + threadIdx.x;           // i = d*768 + h
  if (i < Hdim * Hdim) {
    int d = i / Hdim, h = i - d * Hdim;
    w1t[i] = (_Float16)Wp[h * Hdim + d];
  }
}

// wot[label][h] = W_out[h][label]
__global__ void pack_wot(const float* __restrict__ Wo, _Float16* __restrict__ wot) {
  int i = blockIdx.x * blockDim.x + threadIdx.x;           // i = label*768 + h
  if (i < NLAB * Hdim) {
    int l = i / Hdim, h = i - l * Hdim;
    wot[i] = (_Float16)Wo[h * NLAB + l];
  }
}

// XI[bl][d] = sum_h x[bl][h]*W2[h][d] + b_proj[d] ; XJ[bl][d] = sum_h x[bl][h]*W3[h][d]
__global__ void xiw_kernel(const float* __restrict__ x, const float* __restrict__ Wp,
                           const float* __restrict__ bp,
                           float* __restrict__ XI, float* __restrict__ XJ) {
  int bl = blockIdx.y;
  int d  = blockIdx.x * blockDim.x + threadIdx.x;
  const float* xr = x + bl * Hdim;
  const float* w2 = Wp + Hdim * Hdim;        // W2 rows
  const float* w3 = Wp + 2 * Hdim * Hdim;    // W3 rows
  float ai = 0.f, aj = 0.f;
  for (int h = 0; h < Hdim; ++h) {
    float xv = xr[h];
    ai = __builtin_fmaf(xv, w2[h * Hdim + d], ai);
    aj = __builtin_fmaf(xv, w3[h * Hdim + d], aj);
  }
  XI[bl * Hdim + d] = ai + bp[d];
  XJ[bl * Hdim + d] = aj;
}

// ---------------- main fused kernel ----------------
// block = 128 threads (4 waves); each wave owns one tile: fixed i, 16 consecutive j,
// with an N=64 register tile over d. grid = 4*256*16 / 4 = 4096 blocks.

__global__ __launch_bounds__(128)
void biaffine_kernel(const _Float16* __restrict__ xh,
                     const _Float16* __restrict__ w1t,
                     const _Float16* __restrict__ wot,
                     const float* __restrict__ XI,
                     const float* __restrict__ XJ,
                     const float* __restrict__ b_out,
                     float* __restrict__ out) {
  __shared__ __align__(16) _Float16 Pt[4][16 * 64];   // per-wave tanh tile (m x 64 d)
  __shared__ __align__(16) float    Ot[4][NLAB * 16]; // per-wave out tile  (label x j)

  const int lane = threadIdx.x & 31;
  const int wv   = threadIdx.x >> 5;
  const int half = lane >> 4;      // 0/1
  const int n    = lane & 15;      // N index (col) / M index (A rows)

  const int wglob = blockIdx.x * 4 + wv;       // 0..16383
  const int bidx  = wglob >> 12;               // 4096 tiles per batch
  const int rem   = wglob & 4095;
  const int i     = rem >> 4;                  // 0..255
  const int j0    = (rem & 15) << 4;           // 0..240 step 16

  // A-fragment source rows (f16). A[m][k] = x_i[k] * x_{j0+m}[k]; M = lane&15.
  const _Float16* xi_row = xh + (size_t)(bidx * Lseq + i) * Hdim + half * 8;
  const _Float16* xj_row = xh + (size_t)(bidx * Lseq + j0 + n) * Hdim + half * 8;
  // B-fragment rows for GEMM1: w1t[d0+n][k0 + half*8 + {0..7, 16..23}]
  const _Float16* w1row = w1t + (size_t)n * Hdim + half * 8;  // + d0*Hdim + k0 later

  const float* XIrow = XI + (size_t)(bidx * Lseq + i) * Hdim;
  const float* XJrow = XJ + (size_t)(bidx * Lseq + j0) * Hdim;

  v8f acc2[4];
  #pragma unroll
  for (int g = 0; g < 4; ++g) acc2[g] = (v8f){};

  for (int dp = 0; dp < Hdim / 64; ++dp) {        // 12 outer panels of 64 d's
    const int dbase = dp * 64;
    v8f acc[4];
    #pragma unroll
    for (int t = 0; t < 4; ++t) acc[t] = (v8f){};

    const _Float16* bpan = w1row + (size_t)dbase * Hdim;
    #pragma unroll 2
    for (int kc = 0; kc < Hdim / 32; ++kc) {
      const int k0 = kc * 32;
      AF ai, aj, a;
      ai.h[0] = ld8(xi_row + k0);   ai.h[1] = ld8(xi_row + k0 + 16);
      aj.h[0] = ld8(xj_row + k0);   aj.h[1] = ld8(xj_row + k0 + 16);
      a.v = ai.v * aj.v;                               // v_pk_mul_f16 x8, reused 4x
      #pragma unroll
      for (int t = 0; t < 4; ++t) {
        AF bw;
        const _Float16* bb = bpan + (size_t)(t * 16) * Hdim + k0;
        bw.h[0] = ld8(bb);  bw.h[1] = ld8(bb + 16);
        acc[t] = __builtin_amdgcn_wmma_f32_16x16x32_f16(
            false, a.v, false, bw.v, (short)0, acc[t], false, false);
      }
    }

    // epilogue: + XI(i,d) + XJ(j,d), tanh, stash as f16 in LDS (m x 64 layout)
    #pragma unroll
    for (int t = 0; t < 4; ++t) {
      const int d0 = dbase + t * 16;
      const float xiv = XIrow[d0 + n];
      #pragma unroll
      for (int v = 0; v < 8; ++v) {
        const int m = v + 8 * half;                   // D row = vgpr + 8*half
        float tv = ftanh(acc[t][v] + xiv + XJrow[(size_t)m * Hdim + d0 + n]);
        Pt[wv][m * 64 + t * 16 + n] = (_Float16)tv;
      }
    }

    // wave-internal D->A layout change through LDS (DS is in-order per wave)
    asm volatile("s_wait_dscnt 0" ::: "memory");
    #pragma unroll
    for (int s = 0; s < 2; ++s) {                     // two K2-chunks of 32
      AF a2;
      const _Float16* pb = &Pt[wv][(lane & 15) * 64 + s * 32 + half * 8];
      a2.h[0] = *(const v8h*)pb;
      a2.h[1] = *(const v8h*)(pb + 16);
      #pragma unroll
      for (int g = 0; g < 4; ++g) {
        AF b2;
        const _Float16* wb = wot + (size_t)(g * 16 + n) * Hdim + dbase + s * 32 + half * 8;
        b2.h[0] = ld8(wb);  b2.h[1] = ld8(wb + 16);
        acc2[g] = __builtin_amdgcn_wmma_f32_16x16x32_f16(
            false, a2.v, false, b2.v, (short)0, acc2[g], false, false);
      }
    }
  }

  // transpose 16x64 result through LDS -> coalesced (B,64,L,L) stores
  #pragma unroll
  for (int g = 0; g < 4; ++g)
    #pragma unroll
    for (int v = 0; v < 8; ++v)
      Ot[wv][(g * 16 + n) * 16 + v + 8 * half] = acc2[g][v];

  asm volatile("s_wait_dscnt 0" ::: "memory");

  float* obase = out + (size_t)bidx * NLAB * Lseq * Lseq + (size_t)i * Lseq + j0;
  #pragma unroll
  for (int it = 0; it < 8; ++it) {
    const int c = it * 32 + lane;        // 0..255 float4 chunks
    const int label = c >> 2;
    const int q = c & 3;
    float4 val = *(const float4*)&Ot[wv][label * 16 + q * 4];
    const float bo = b_out[label];
    val.x += bo; val.y += bo; val.z += bo; val.w += bo;
    *(float4*)(obase + (size_t)label * Lseq * Lseq + q * 4) = val;
  }
}

// ---------------- launcher ----------------

extern "C" void kernel_launch(void* const* d_in, const int* in_sizes, int n_in,
                              void* d_out, int out_size, void* d_ws, size_t ws_size,
                              hipStream_t stream) {
  const float* x   = (const float*)d_in[0];   // (4,256,768)
  const float* Wp  = (const float*)d_in[1];   // (2304,768)
  const float* bp  = (const float*)d_in[2];   // (768,)
  const float* Wo  = (const float*)d_in[3];   // (768,64)
  const float* bo  = (const float*)d_in[4];   // (64,)
  float* out = (float*)d_out;                 // (4,64,256,256)

  // workspace carve-up (~9.2 MB total)
  char* ws = (char*)d_ws;
  _Float16* xh  = (_Float16*)(ws);                       // 1,572,864 B
  _Float16* w1t = (_Float16*)(ws + 1572864);             // 1,179,648 B
  _Float16* wot = (_Float16*)(ws + 2752512);             //    98,304 B
  float*    XI  = (float*)   (ws + 2850816);             // 3,145,728 B
  float*    XJ  = (float*)   (ws + 5996544);             // 3,145,728 B

  const int NX = Bsz * Lseq * Hdim;                      // 786432
  pack_xh <<<(NX + 255) / 256, 256, 0, stream>>>(x, xh, NX);
  pack_w1t<<<(Hdim * Hdim + 255) / 256, 256, 0, stream>>>(Wp, w1t);
  pack_wot<<<(NLAB * Hdim + 255) / 256, 256, 0, stream>>>(Wo, wot);
  xiw_kernel<<<dim3(Hdim / 256, Bsz * Lseq), 256, 0, stream>>>(x, Wp, bp, XI, XJ);

  // 16384 wave-tiles, 4 waves per block
  biaffine_kernel<<<4096, 128, 0, stream>>>(xh, w1t, wot, XI, XJ, bo, out);
}
